// UVMSR_60112362275365
// MI455X (gfx1250) — matrix operand order repaired and checked by reading, not compiled, not run on hardware
//
#include <hip/hip_runtime.h>
#include <hip/hip_bf16.h>
#include <math.h>

typedef __attribute__((ext_vector_type(16))) _Float16 v16h;
typedef __attribute__((ext_vector_type(8)))  float    v8f;

#define BB 2
#define LL 16384
#define DM 64
#define ED 128
#define NS 16
#define BL (BB*LL)
#define DBC_LD 48
#define NCH 128
#define CHL 128   // NCH*CHL == LL

// ---------------- WMMA fragment helpers (CDNA5 16x16x32 f16 layouts) ----------------
// A (16x32, MxK): lanes 0-15 row M=lane, element e -> K = e + (e&8) (+kb 0)
//                 lanes 16-31 row M=lane-16, element e -> K = e + 8 + (e&8)
__device__ __forceinline__ v16h frag_a_lds(const _Float16* A, int ld, int r0, int k0) {
  int lane = threadIdx.x & 31;
  int row  = r0 + (lane & 15);
  int kb   = k0 + ((lane & 16) ? 8 : 0);
  v16h a;
#pragma unroll
  for (int e = 0; e < 16; ++e) a[e] = A[row * ld + kb + e + (e & 8)];
  return a;
}

// B (32x16, KxN) built from W stored row-major [N][K]: B[k][n] = W[n][k]
// lanes 0-15: N=lane, element e -> K = k0+e+(e&8) ; lanes 16-31: N=lane-16, K = k0+8+e+(e&8)
__device__ __forceinline__ v16h frag_b_lds(const _Float16* W, int ld, int n0, int k0) {
  int lane = threadIdx.x & 31;
  int n    = n0 + (lane & 15);
  int kb   = k0 + ((lane & 16) ? 8 : 0);
  v16h b;
#pragma unroll
  for (int e = 0; e < 16; ++e) b[e] = W[n * ld + kb + e + (e & 8)];
  return b;
}

// ---------------- Kernel 1: RMSNorm + in_proj (M=BL, K=64, N=256) ----------------
__global__ void k_rmsnorm_inproj(const float* __restrict__ x, const float* __restrict__ norm_w,
                                 const float* __restrict__ w, float* __restrict__ xi,
                                 float* __restrict__ z) {
  __shared__ _Float16 As[64 * 64];    // 64 rows x K=64
  __shared__ _Float16 Ws[256 * 64];   // W[n][k]
  int tid  = threadIdx.x;
  int row0 = blockIdx.x * 64;
  for (int i = tid; i < 256 * 64; i += 256) Ws[i] = (_Float16)w[i];
  if (tid < 64) {
    const float* xr = x + (size_t)(row0 + tid) * DM;
    float s = 0.f;
    for (int j = 0; j < DM; ++j) { float v = xr[j]; s += v * v; }
    float sc = rsqrtf(s * (1.f / DM) + 1e-5f);
    for (int j = 0; j < DM; ++j) As[tid * 64 + j] = (_Float16)(xr[j] * sc * norm_w[j]);
  }
  __syncthreads();
  int wave = tid >> 5;
  for (int ci = 0; ci < 2; ++ci) {
    int ct = wave * 2 + ci;                       // 16 col tiles over 8 waves
    float* dst = (ct < 8) ? xi : z;
    int c0 = (ct < 8) ? ct * 16 : (ct - 8) * 16;
    for (int rt = 0; rt < 4; ++rt) {
      v8f acc = {};
#pragma unroll
      for (int ks = 0; ks < 64; ks += 32) {
        v16h a = frag_a_lds(As, 64, rt * 16, ks);
        v16h b = frag_b_lds(Ws, 64, ct * 16, ks);
        acc = __builtin_amdgcn_wmma_f32_16x16x32_f16(false, a, false, b, (short)0, acc, false, false);
      }
      int lane = tid & 31;
      int col  = c0 + (lane & 15);
      int r    = row0 + rt * 16 + ((lane & 16) ? 8 : 0);
#pragma unroll
      for (int i = 0; i < 8; ++i) dst[(size_t)(r + i) * ED + col] = acc[i];
    }
  }
}

// ---------------- Kernel 2: both causal depthwise convs ----------------
__global__ void k_dwconv(const float* __restrict__ xi,
                         const float* __restrict__ wf, const float* __restrict__ bf,
                         const float* __restrict__ wb, const float* __restrict__ bb,
                         float* __restrict__ uf, float* __restrict__ ub) {
  int idx = blockIdx.x * blockDim.x + threadIdx.x;
  if (idx >= BL * ED) return;
  int e   = idx & (ED - 1);
  int tok = idx >> 7;
  int l   = tok & (LL - 1);
  float af = bf[e], ab = bb[e];
#pragma unroll
  for (int k = 0; k < 4; ++k) {
    int ll = l - 3 + k;
    if (ll >= 0) {
      float v = xi[(size_t)(tok - (l - ll)) * ED + e];
      af += wf[e * 4 + k] * v;
      ab += wb[e * 4 + k] * v;
    }
  }
  uf[idx] = af;
  ub[idx] = ab;
}

// ---------------- Kernel 3: xproj GEMM (M=BL, K=128, N=36 padded to 48) ----------------
__global__ void k_xproj(const float* __restrict__ u, const float* __restrict__ w,
                        float* __restrict__ dbc) {
  __shared__ _Float16 Us[128 * 128];
  __shared__ _Float16 Wx[48 * 128];
  int tid  = threadIdx.x;
  int row0 = blockIdx.x * 128;
  for (int i = tid; i < 48 * 128; i += 256) {
    int n = i >> 7, k = i & 127;
    Wx[i] = (n < 36) ? (_Float16)w[n * 128 + k] : (_Float16)0.f;
  }
  for (int i = tid; i < 128 * 128; i += 256) Us[i] = (_Float16)u[(size_t)row0 * ED + i];
  __syncthreads();
  int rt = tid >> 5;                              // wave -> row tile (8 tiles)
  for (int ct = 0; ct < 3; ++ct) {
    v8f acc = {};
#pragma unroll
    for (int ks = 0; ks < 128; ks += 32) {
      v16h a = frag_a_lds(Us, 128, rt * 16, ks);
      v16h b = frag_b_lds(Wx, 128, ct * 16, ks);
      acc = __builtin_amdgcn_wmma_f32_16x16x32_f16(false, a, false, b, (short)0, acc, false, false);
    }
    int lane = tid & 31;
    int col  = ct * 16 + (lane & 15);
    int r    = row0 + rt * 16 + ((lane & 16) ? 8 : 0);
#pragma unroll
    for (int i = 0; i < 8; ++i) dbc[(size_t)(r + i) * DBC_LD + col] = acc[i];
  }
}

// ---------------- Kernel 3b: delta = softplus(dlt @ dtproj_w.T + b) ----------------
__global__ void k_delta(const float* __restrict__ dbc, const float* __restrict__ dtw,
                        const float* __restrict__ dtb, float* __restrict__ delta) {
  int idx = blockIdx.x * blockDim.x + threadIdx.x;
  if (idx >= BL * ED) return;
  int e   = idx & (ED - 1);
  int tok = idx >> 7;
  float d = dtb[e];
#pragma unroll
  for (int r = 0; r < 4; ++r) d += dbc[(size_t)tok * DBC_LD + r] * dtw[e * 4 + r];
  delta[idx] = (d > 20.f) ? d : log1pf(expf(d));
}

// ---------------- Scan phase 1: per-chunk (prod a, partial h) ----------------
__global__ void k_scan1(const float* __restrict__ delta, const float* __restrict__ u,
                        const float* __restrict__ dbc, const float* __restrict__ A_log,
                        float* __restrict__ csa, float* __restrict__ csh) {
  int seq   = threadIdx.x;        // b*ED + e  (256 sequences)
  int chunk = blockIdx.x;
  int b = seq >> 7, e = seq & (ED - 1);
  float An[NS], ap[NS], h[NS];
#pragma unroll
  for (int n = 0; n < NS; ++n) { An[n] = -expf(A_log[e * NS + n]); ap[n] = 1.f; h[n] = 0.f; }
  int l0 = chunk * CHL;
  for (int l = l0; l < l0 + CHL; ++l) {
    size_t tok = (size_t)b * LL + l;
    float dl = delta[tok * ED + e];
    float du = dl * u[tok * ED + e];
    const float* brow = dbc + tok * DBC_LD + 4;
#pragma unroll
    for (int n = 0; n < NS; ++n) {
      float a = expf(dl * An[n]);
      h[n]  = a * h[n] + du * brow[n];
      ap[n] *= a;
    }
  }
  int base = (seq * NCH + chunk) * NS;
#pragma unroll
  for (int n = 0; n < NS; ++n) { csa[base + n] = ap[n]; csh[base + n] = h[n]; }
}

// ---------------- Scan phase 2: sequential chunk-prefix combine ----------------
__global__ void k_scan2(const float* __restrict__ csa, const float* __restrict__ csh,
                        float* __restrict__ h0) {
  int seq = threadIdx.x;          // 256 sequences, 1 block
  float hc[NS];
#pragma unroll
  for (int n = 0; n < NS; ++n) hc[n] = 0.f;
  for (int c = 0; c < NCH; ++c) {
    int base = (seq * NCH + c) * NS;
#pragma unroll
    for (int n = 0; n < NS; ++n) {
      h0[base + n] = hc[n];
      hc[n] = csa[base + n] * hc[n] + csh[base + n];
    }
  }
}

// ---------------- Scan phase 3: apply with prefix, emit y (+= for dir 1) ----------------
__global__ void k_scan3(const float* __restrict__ delta, const float* __restrict__ u,
                        const float* __restrict__ dbc, const float* __restrict__ A_log,
                        const float* __restrict__ Dp, const float* __restrict__ h0,
                        float* __restrict__ y, int accumulate) {
  int seq   = threadIdx.x;
  int chunk = blockIdx.x;
  int b = seq >> 7, e = seq & (ED - 1);
  float An[NS], h[NS];
  int base = (seq * NCH + chunk) * NS;
#pragma unroll
  for (int n = 0; n < NS; ++n) { An[n] = -expf(A_log[e * NS + n]); h[n] = h0[base + n]; }
  float dpe = Dp[e];
  int l0 = chunk * CHL;
  for (int l = l0; l < l0 + CHL; ++l) {
    size_t tok = (size_t)b * LL + l;
    float dl = delta[tok * ED + e];
    float uu = u[tok * ED + e];
    float du = dl * uu;
    const float* brow = dbc + tok * DBC_LD + 4;
    const float* crow = dbc + tok * DBC_LD + 20;
    float yv = dpe * uu;
#pragma unroll
    for (int n = 0; n < NS; ++n) {
      float a = expf(dl * An[n]);
      h[n] = a * h[n] + du * brow[n];
      yv  += h[n] * crow[n];
    }
    if (accumulate) y[tok * ED + e] += yv;
    else            y[tok * ED + e]  = yv;
  }
}

// ---------------- Kernel 5: gate with mish(z), out_proj GEMM (K=128,N=64), +skip ----------------
__global__ void k_out(const float* __restrict__ y, const float* __restrict__ z,
                      const float* __restrict__ w, const float* __restrict__ x,
                      float* __restrict__ out) {
  __shared__ _Float16 Ys[64 * 128];
  __shared__ _Float16 Wo[64 * 128];
  int tid  = threadIdx.x;
  int row0 = blockIdx.x * 64;
  for (int i = tid; i < 64 * 128; i += 256) Wo[i] = (_Float16)w[i];
  for (int i = tid; i < 64 * 128; i += 256) {
    size_t gi = (size_t)row0 * ED + i;
    float zv = z[gi];
    float sp = (zv > 20.f) ? zv : log1pf(expf(zv));
    Ys[i] = (_Float16)(y[gi] * zv * tanhf(sp));
  }
  __syncthreads();
  int wave = tid >> 5;
  for (int ti = 0; ti < 2; ++ti) {
    int t  = wave * 2 + ti;                       // 4 row tiles x 4 col tiles
    int rt = t >> 2, ct = t & 3;
    v8f acc = {};
#pragma unroll
    for (int ks = 0; ks < 128; ks += 32) {
      v16h a = frag_a_lds(Ys, 128, rt * 16, ks);
      v16h b = frag_b_lds(Wo, 128, ct * 16, ks);
      acc = __builtin_amdgcn_wmma_f32_16x16x32_f16(false, a, false, b, (short)0, acc, false, false);
    }
    int lane = tid & 31;
    int col  = ct * 16 + (lane & 15);
    int r    = row0 + rt * 16 + ((lane & 16) ? 8 : 0);
#pragma unroll
    for (int i = 0; i < 8; ++i)
      out[(size_t)(r + i) * DM + col] = acc[i] + x[(size_t)(r + i) * DM + col];
  }
}

extern "C" void kernel_launch(void* const* d_in, const int* in_sizes, int n_in,
                              void* d_out, int out_size, void* d_ws, size_t ws_size,
                              hipStream_t stream) {
  const float* x        = (const float*)d_in[0];
  const float* norm_w   = (const float*)d_in[1];
  const float* in_proj  = (const float*)d_in[2];
  const float* convf_w  = (const float*)d_in[3];
  const float* convf_b  = (const float*)d_in[4];
  const float* convb_w  = (const float*)d_in[5];
  const float* convb_b  = (const float*)d_in[6];
  const float* xproj_w  = (const float*)d_in[7];
  const float* dtproj_w = (const float*)d_in[8];
  const float* dtproj_b = (const float*)d_in[9];
  const float* A_log    = (const float*)d_in[10];
  const float* Dp       = (const float*)d_in[11];
  const float* out_proj = (const float*)d_in[12];
  float* out = (float*)d_out;

  // workspace layout (floats): ~24.1M floats ≈ 96 MB
  float* ws    = (float*)d_ws;
  float* xi    = ws;                         // BL*ED ; reused as y after conv
  float* z     = xi    + (size_t)BL * ED;
  float* uf    = z     + (size_t)BL * ED;
  float* ub    = uf    + (size_t)BL * ED;
  float* delta = ub    + (size_t)BL * ED;    // reused per direction
  float* dbc   = delta + (size_t)BL * ED;    // BL*48, reused per direction
  float* csa   = dbc   + (size_t)BL * DBC_LD;
  float* csh   = csa   + (size_t)BB * ED * NCH * NS;
  float* h0    = csh   + (size_t)BB * ED * NCH * NS;

  k_rmsnorm_inproj<<<BL / 64, 256, 0, stream>>>(x, norm_w, in_proj, xi, z);
  k_dwconv<<<(BL * ED) / 256, 256, 0, stream>>>(xi, convf_w, convf_b, convb_w, convb_b, uf, ub);

  for (int dir = 0; dir < 2; ++dir) {
    const float* u = dir ? ub : uf;
    k_xproj<<<BL / 128, 256, 0, stream>>>(u, xproj_w, dbc);
    k_delta<<<(BL * ED) / 256, 256, 0, stream>>>(dbc, dtproj_w, dtproj_b, delta);
    k_scan1<<<NCH, 256, 0, stream>>>(delta, u, dbc, A_log, csa, csh);
    k_scan2<<<1, 256, 0, stream>>>(csa, csh, h0);
    k_scan3<<<NCH, 256, 0, stream>>>(delta, u, dbc, A_log, Dp, h0, xi /*y*/, dir);
  }

  k_out<<<BL / 64, 256, 0, stream>>>(xi /*y*/, z, out_proj, x, out);
}
